// AttentionHead_72662256714554
// MI455X (gfx1250) — matrix operand order repaired and checked by reading, not compiled, not run on hardware
//
#include <hip/hip_runtime.h>
#include <hip/hip_bf16.h>

typedef _Float16 v16h __attribute__((ext_vector_type(16)));
typedef _Float16 v8h  __attribute__((ext_vector_type(8)));
typedef float    v8f  __attribute__((ext_vector_type(8)));

#define BT  2048
#define CC  768
#define HD  64
#define NB  8

static __device__ inline v8f wmma16(v16h a, v16h b, v8f c) {
    return __builtin_amdgcn_wmma_f32_16x16x32_f16(
        /*neg_a=*/false, a, /*neg_b=*/false, b,
        /*c_mod=*/(short)0, c, /*reuse_a=*/false, /*reuse_b=*/false);
}

static __device__ inline v16h cat8(v8h lo, v8h hi) {
    return __builtin_shufflevector(lo, hi, 0,1,2,3,4,5,6,7,8,9,10,11,12,13,14,15);
}

// ---- CDNA5 async global->LDS copy (ASYNCcnt path) --------------------------
// VDST = per-lane LDS byte offset, VADDR = 64-bit global address (GV mode).
static __device__ inline void async_copy_b128(const void* gsrc, void* ldst) {
    unsigned lds = (unsigned)(size_t)(__attribute__((address_space(3))) void*)ldst;
    unsigned long long ga = (unsigned long long)(size_t)gsrc;
    asm volatile("global_load_async_to_lds_b128 %0, %1, off"
                 :: "v"(lds), "v"(ga) : "memory");
}

static __device__ inline void wait_async0() {
#if defined(__has_builtin) && __has_builtin(__builtin_amdgcn_s_wait_asynccnt)
    __builtin_amdgcn_s_wait_asynccnt(0);
#else
    asm volatile("s_wait_asynccnt 0x0" ::: "memory");
#endif
}

// ---------------------------------------------------------------------------
// Kernel 0: W [768][64] f32 -> Wt [3][64][768] f16 (scale 1/8 folded into Wq)
// ---------------------------------------------------------------------------
__global__ __launch_bounds__(256) void wconv_kernel(
    const float* __restrict__ wq, const float* __restrict__ wk,
    const float* __restrict__ wv, _Float16* __restrict__ wt)
{
    int idx = blockIdx.x * 256 + threadIdx.x;
    if (idx >= 3 * HD * CC) return;
    int z   = idx / (HD * CC);
    int rem = idx % (HD * CC);
    int nn  = rem / CC;           // output column 0..63
    int cc  = rem % CC;           // input channel 0..767
    const float* W = (z == 0) ? wq : (z == 1) ? wk : wv;
    float v = W[(size_t)cc * HD + nn];
    if (z == 0) v *= 0.125f;      // 1/sqrt(64) folded into Wq
    wt[idx] = (_Float16)v;
}

// ---------------------------------------------------------------------------
// Kernel 1: q/k/v = x @ W   ->  f16 [B*T][64]
// block = 256 (8 waves); wave owns a 16-row x 64-col output tile
// grid = (128 row-blocks, 3 matrices)
// ---------------------------------------------------------------------------
__global__ __launch_bounds__(256) void qkv_proj_kernel(
    const float* __restrict__ x, const _Float16* __restrict__ wt,
    _Float16* __restrict__ qh, _Float16* __restrict__ kh, _Float16* __restrict__ vh)
{
    const int tid  = threadIdx.x;
    const int wv   = tid >> 5;
    const int lane = tid & 31;
    const int n    = lane & 15;     // column within 16 / A-row
    const int g    = lane >> 4;     // lane half-group
    const int z    = blockIdx.y;
    const int row0 = blockIdx.x * 128 + wv * 16;

    const _Float16* w = wt + (size_t)z * HD * CC;
    _Float16* outp = (z == 0) ? qh : (z == 1) ? kh : vh;

    const float* xr = x + (size_t)(row0 + n) * CC;

    v8f c0 = {}, c1 = {}, c2 = {}, c3 = {};

    for (int k0 = 0; k0 < CC; k0 += 32) {
        // A fragment (16x32 f16): lane holds row n, K = 16g + s (contiguous)
        float4 f0 = *(const float4*)(xr + k0 + g * 16);
        float4 f1 = *(const float4*)(xr + k0 + g * 16 + 4);
        float4 f2 = *(const float4*)(xr + k0 + g * 16 + 8);
        float4 f3 = *(const float4*)(xr + k0 + g * 16 + 12);
        v16h a;
        a[0]=(_Float16)f0.x; a[1]=(_Float16)f0.y; a[2]=(_Float16)f0.z; a[3]=(_Float16)f0.w;
        a[4]=(_Float16)f1.x; a[5]=(_Float16)f1.y; a[6]=(_Float16)f1.z; a[7]=(_Float16)f1.w;
        a[8]=(_Float16)f2.x; a[9]=(_Float16)f2.y; a[10]=(_Float16)f2.z; a[11]=(_Float16)f2.w;
        a[12]=(_Float16)f3.x; a[13]=(_Float16)f3.y; a[14]=(_Float16)f3.z; a[15]=(_Float16)f3.w;

        // B fragments (32x16 f16) from transposed weights: contiguous 32B
        v16h b0 = *(const v16h*)(w + (size_t)( 0 + n) * CC + k0 + g * 16);
        v16h b1 = *(const v16h*)(w + (size_t)(16 + n) * CC + k0 + g * 16);
        v16h b2 = *(const v16h*)(w + (size_t)(32 + n) * CC + k0 + g * 16);
        v16h b3 = *(const v16h*)(w + (size_t)(48 + n) * CC + k0 + g * 16);

        c0 = wmma16(a, b0, c0);
        c1 = wmma16(a, b1, c1);
        c2 = wmma16(a, b2, c2);
        c3 = wmma16(a, b3, c3);
    }

    // store f16: C/D layout row M = 8g+i, col N = n
    #pragma unroll
    for (int i = 0; i < 8; ++i) {
        int r = row0 + 8 * g + i;
        _Float16* orow = outp + (size_t)r * HD;
        orow[ 0 + n] = (_Float16)c0[i];
        orow[16 + n] = (_Float16)c1[i];
        orow[32 + n] = (_Float16)c2[i];
        orow[48 + n] = (_Float16)c3[i];
    }
}

// ---------------------------------------------------------------------------
// Kernel 2: flash attention with causal mask + relative position bias
// grid = (16 q-blocks, 8 batches), block = 256 (8 waves x 16 q-rows)
// key tile = 32; K staged via async global->LDS; V transposed into LDS;
// online softmax; all GEMMs via v_wmma_f32_16x16x32_f16
// ---------------------------------------------------------------------------
#define VT_STRIDE 48   // halves; row stride 96B keeps 32B alignment for v16h loads

__global__ __launch_bounds__(256) void fattn_kernel(
    const _Float16* __restrict__ qh, const _Float16* __restrict__ kh,
    const _Float16* __restrict__ vh, const float* __restrict__ bias_table,
    float* __restrict__ out)
{
    __shared__ float    bias_s[BT];                 // 8 KB
    __shared__ _Float16 kt[32 * HD];                // 4 KB  raw K tile [key][dim]
    __shared__ _Float16 vt[HD * VT_STRIDE];         // 6 KB  transposed V [dim][key]
    __shared__ _Float16 pt[8][16 * 32];             // 8 KB  per-wave P tile

    const int tid  = threadIdx.x;
    const int wv   = tid >> 5;
    const int lane = tid & 31;
    const int n    = lane & 15;
    const int g    = lane >> 4;
    const int b    = blockIdx.y;
    const int qblk = blockIdx.x;
    const int qw   = qblk * 128 + wv * 16;          // wave's first query row

    for (int i = tid; i < BT; i += 256) bias_s[i] = bias_table[i];

    const _Float16* qb_ = qh + (size_t)b * BT * HD;
    const _Float16* kb_ = kh + (size_t)b * BT * HD;
    const _Float16* vb_ = vh + (size_t)b * BT * HD;

    // Q A-fragments (16x32 each, two chunks cover head dim 64); scale pre-folded
    v16h aq0, aq1;
    {
        const _Float16* qr = qb_ + (size_t)(qw + n) * HD;
        v8h l0 = *(const v8h*)(qr +  0 + g * 16);
        v8h h0 = *(const v8h*)(qr +  8 + g * 16);
        v8h l1 = *(const v8h*)(qr + 32 + g * 16);
        v8h h1 = *(const v8h*)(qr + 40 + g * 16);
        aq0 = cat8(l0, h0);
        aq1 = cat8(l1, h1);
    }

    float m[8], l[8];
    v8f o0 = {}, o1 = {}, o2 = {}, o3 = {};
    #pragma unroll
    for (int i = 0; i < 8; ++i) { m[i] = -1e30f; l[i] = 0.0f; }

    const int tmax = qblk * 4 + 3;                  // last key tile needed by block
    for (int t = 0; t <= tmax; ++t) {
        __syncthreads();

        // async-copy K tile (32 keys x 64 dims f16 = 4KB) into LDS:
        // 256 threads x one b128 transfer each; tracked by ASYNCcnt
        {
            int key = tid >> 3;            // 0..31
            int seg = (tid & 7) * 8;       // halves 0,8,...,56
            async_copy_b128(kb_ + (size_t)(t * 32 + key) * HD + seg,
                            &kt[key * HD + seg]);
        }

        // cooperative transposed V tile load: vt[dim][key], keys t*32..t*32+31
        {
            int key = tid & 31;
            int d0  = (tid >> 5) * 8;
            v8h vrow = *(const v8h*)(vb_ + (size_t)(t * 32 + key) * HD + d0);
            #pragma unroll
            for (int j = 0; j < 8; ++j) vt[(d0 + j) * VT_STRIDE + key] = vrow[j];
        }
        if (t < tmax) {   // prefetch next V tile (global_prefetch_b8)
            __builtin_prefetch(vb_ + (size_t)((t + 1) * 32 + lane) * HD, 0, 0);
        }
        wait_async0();       // this wave's async K transfers complete
        __syncthreads();     // ... and everyone else's

        // S = Q K^T : two 16x16 sub-tiles (keys n and 16+n), two 32-dim chunks;
        // B fragments now come from the LDS-staged K tile
        v8f s0 = {}, s1 = {};
        {
            const _Float16* k0p = &kt[( 0 + n) * HD];
            const _Float16* k1p = &kt[(16 + n) * HD];
            v16h kb00 = *(const v16h*)(k0p +  0 + g * 16);
            v16h kb01 = *(const v16h*)(k0p + 32 + g * 16);
            v16h kb10 = *(const v16h*)(k1p +  0 + g * 16);
            v16h kb11 = *(const v16h*)(k1p + 32 + g * 16);
            s0 = wmma16(aq0, kb00, s0);
            s0 = wmma16(aq1, kb01, s0);
            s1 = wmma16(aq0, kb10, s1);
            s1 = wmma16(aq1, kb11, s1);
        }

        // bias + causal mask, online softmax update
        #pragma unroll
        for (int i = 0; i < 8; ++i) {
            int qa  = qw + 8 * g + i;
            int k0a = t * 32 + n;
            int k1a = k0a + 16;
            float v0 = (k0a <= qa) ? (s0[i] + bias_s[qa - k0a]) : -1e30f;
            float v1 = (k1a <= qa) ? (s1[i] + bias_s[qa - k1a]) : -1e30f;

            float mt = fmaxf(v0, v1);
            mt = fmaxf(mt, __shfl_xor(mt, 1, 32));
            mt = fmaxf(mt, __shfl_xor(mt, 2, 32));
            mt = fmaxf(mt, __shfl_xor(mt, 4, 32));
            mt = fmaxf(mt, __shfl_xor(mt, 8, 32));
            float mn   = fmaxf(m[i], mt);
            float p0   = __expf(v0 - mn);
            float p1   = __expf(v1 - mn);
            float corr = __expf(m[i] - mn);
            float rs   = p0 + p1;
            rs += __shfl_xor(rs, 1, 32);
            rs += __shfl_xor(rs, 2, 32);
            rs += __shfl_xor(rs, 4, 32);
            rs += __shfl_xor(rs, 8, 32);
            l[i] = l[i] * corr + rs;
            m[i] = mn;
            o0[i] *= corr; o1[i] *= corr; o2[i] *= corr; o3[i] *= corr;

            _Float16* pr = &pt[wv][(8 * g + i) * 32];
            pr[n]      = (_Float16)p0;
            pr[16 + n] = (_Float16)p1;
        }
        __syncthreads();   // P tile visible across lanes of the wave

        // P as A-fragment (16x32), V B-fragments from transposed LDS tile
        const _Float16* pw = &pt[wv][0];
        v8h plo = *(const v8h*)(pw + n * 32 + g * 16);
        v8h phi = *(const v8h*)(pw + n * 32 + 8 + g * 16);
        v16h pa = cat8(plo, phi);

        v16h vb0 = *(const v16h*)(&vt[( 0 + n) * VT_STRIDE + g * 16]);
        v16h vb1 = *(const v16h*)(&vt[(16 + n) * VT_STRIDE + g * 16]);
        v16h vb2 = *(const v16h*)(&vt[(32 + n) * VT_STRIDE + g * 16]);
        v16h vb3 = *(const v16h*)(&vt[(48 + n) * VT_STRIDE + g * 16]);
        o0 = wmma16(pa, vb0, o0);
        o1 = wmma16(pa, vb1, o1);
        o2 = wmma16(pa, vb2, o2);
        o3 = wmma16(pa, vb3, o3);
    }

    // epilogue: normalize and store f32 output [B][T][64]
    #pragma unroll
    for (int i = 0; i < 8; ++i) {
        float inv = 1.0f / l[i];
        int qa = qw + 8 * g + i;
        float* orow = out + ((size_t)b * BT + qa) * HD;
        orow[ 0 + n] = o0[i] * inv;
        orow[16 + n] = o1[i] * inv;
        orow[32 + n] = o2[i] * inv;
        orow[48 + n] = o3[i] * inv;
    }
}

// ---------------------------------------------------------------------------
// Launch
// ---------------------------------------------------------------------------
extern "C" void kernel_launch(void* const* d_in, const int* in_sizes, int n_in,
                              void* d_out, int out_size, void* d_ws, size_t ws_size,
                              hipStream_t stream) {
    const float* x    = (const float*)d_in[0];   // [8,2048,768]
    const float* wq   = (const float*)d_in[1];   // [768,64]
    const float* wk   = (const float*)d_in[2];
    const float* wv   = (const float*)d_in[3];
    const float* bias = (const float*)d_in[4];   // [2048,1]
    float* out        = (float*)d_out;           // [8,2048,64]

    // workspace layout (all offsets 256B-aligned)
    const size_t WT_BYTES  = (size_t)3 * HD * CC * sizeof(_Float16);   // 294912
    const size_t QKV_BYTES = (size_t)NB * BT * HD * sizeof(_Float16);  // 2 MB each
    char* ws = (char*)d_ws;
    _Float16* wt = (_Float16*)(ws);
    _Float16* qh = (_Float16*)(ws + WT_BYTES);
    _Float16* kh = (_Float16*)(ws + WT_BYTES + QKV_BYTES);
    _Float16* vh = (_Float16*)(ws + WT_BYTES + 2 * QKV_BYTES);

    wconv_kernel<<<(3 * HD * CC + 255) / 256, 256, 0, stream>>>(wq, wk, wv, wt);
    qkv_proj_kernel<<<dim3(NB * BT / 128, 3), 256, 0, stream>>>(x, wt, qh, kh, vh);
    fattn_kernel<<<dim3(BT / 128, NB), 256, 0, stream>>>(qh, kh, vh, bias, out);
}
